// FodfLoss_73890617360727
// MI455X (gfx1250) — matrix operand order repaired and checked by reading, not compiled, not run on hardware
//
#include <hip/hip_runtime.h>
#include <hip/hip_bf16.h>
#include <math.h>

typedef __attribute__((ext_vector_type(16))) __bf16 v16bf;
typedef __attribute__((ext_vector_type(8)))  float  v8f;

namespace {
constexpr int Bb = 2, Cc = 45, Hh = 160, Ww = 160;
constexpr int HW = Hh * Ww;          // 25600
constexpr int P  = Bb * HW;          // 51200 pixels
constexpr int Vv = 724;
constexpr int VP = 736;              // V padded to 46 tiles of 16
constexpr int NT = VP / 16;          // 46
constexpr int KP = 64;               // K padded for bf16 WMMA: 2 steps of 32
constexpr int WAVES = 8;             // waves per block
constexpr int TILES = P / 16;        // 3200 pixel tiles
constexpr float EPSf = 1e-3f;
}

__global__ void fodf_loss_zero(float* ws) {
  if (threadIdx.x < 16) ws[threadIdx.x] = 0.0f;
}

__global__ __launch_bounds__(256, 1)
void fodf_loss_main(const float* __restrict__ pred,
                    const float* __restrict__ targ,
                    const float* __restrict__ sf,
                    const float* __restrict__ mask,
                    float* __restrict__ ws)
{
  // sf_matrix staged as bf16 in B-operand layout [n][k] (k padded to 64 w/ zeros)
  __shared__ __attribute__((aligned(32))) __bf16 sfB[VP * KP];   // 94208 B
  __shared__ float tileP[WAVES][256];                            // 8 KB
  __shared__ float tileT[WAVES][256];                            // 8 KB

  const int tid  = threadIdx.x;
  const int lane = tid & 31;
  const int wv   = tid >> 5;
  const int half = lane >> 4;      // which half of the wave
  const int col  = lane & 15;      // pixel row (A/M) or direction column (B/N)

  for (int i = tid; i < VP * KP; i += blockDim.x) {
    int n = i >> 6, k = i & (KP - 1);
    float v = (n < Vv && k < Cc) ? sf[k * Vv + n] : 0.0f;
    sfB[i] = (__bf16)v;
  }
  __syncthreads();

  const int tile = blockIdx.x * WAVES + wv;
  const int p0   = tile * 16;
  const int b    = p0 / HW;
  const int q0   = p0 - b * HW;
  const size_t plane = (size_t)HW;
  const size_t base0 = (size_t)b * Cc * plane + (size_t)q0;

  // ---- A fragments (bf16 16x32, contiguous k = 16*half + e, same as B), 2 K-steps ----
  v16bf aP[2], aT[2];
#pragma unroll
  for (int s = 0; s < 2; ++s) {
#pragma unroll
    for (int e = 0; e < 16; ++e) {
      int k = s * 32 + half * 16 + e;
      float pv = 0.0f, tv2 = 0.0f;
      if (k < Cc) {
        size_t off = base0 + (size_t)k * plane + col;
        pv  = pred[off];
        tv2 = targ[off];
      }
      aP[s][e] = (__bf16)pv;
      aT[s][e] = (__bf16)tv2;
    }
  }

  // ---- per-pixel channel statistics: halves split channels, merged by shfl ----
  float psum = 0, tsum = 0, ppv = 0, ttv = 0, ppt = 0, cbsum = 0;
  float bp[5] = {0, 0, 0, 0, 0}, bt[5] = {0, 0, 0, 0, 0};
  {
    const int cs = half ? 23 : 0;
    const int ce = half ? 45 : 23;
    for (int c = cs; c < ce; ++c) {
      size_t off = base0 + (size_t)c * plane + col;
      float pv = pred[off], tv2 = targ[off];
      psum += pv; tsum += tv2;
      ppv  += pv * pv; ttv += tv2 * tv2; ppt += pv * tv2;
      float d = pv - tv2;
      cbsum += sqrtf(d * d + EPSf * EPSf);
      int bi = (c < 1) ? 0 : (c < 6) ? 1 : (c < 15) ? 2 : (c < 28) ? 3 : 4;
#pragma unroll
      for (int q = 0; q < 5; ++q)
        if (bi == q) { bp[q] += pv * pv; bt[q] += tv2 * tv2; }
    }
    psum += __shfl_xor(psum, 16);  tsum += __shfl_xor(tsum, 16);
    ppv  += __shfl_xor(ppv, 16);   ttv  += __shfl_xor(ttv, 16);
    ppt  += __shfl_xor(ppt, 16);   cbsum += __shfl_xor(cbsum, 16);
#pragma unroll
    for (int q = 0; q < 5; ++q) {
      bp[q] += __shfl_xor(bp[q], 16);
      bt[q] += __shfl_xor(bt[q], 16);
    }
  }

  // ---- pass 1: target projection via WMMA, per-pixel top-5 over V ----
  float topv[5] = {-1.f, -1.f, -1.f, -1.f, -1.f};
  int   topi[5] = {-1, -1, -1, -1, -1};
  for (int nt = 0; nt < NT; ++nt) {
    const int n0 = nt * 16;
    v8f acc = {};
#pragma unroll
    for (int s = 0; s < 2; ++s) {
      const __bf16* bsrc = &sfB[(n0 + col) * KP + s * 32 + half * 16];
      v16bf bf = *(const v16bf*)bsrc;
      acc = __builtin_amdgcn_wmma_f32_16x16x32_bf16(
          false, aT[s], false, bf, (short)0, acc, false, false);
    }
#pragma unroll
    for (int r = 0; r < 8; ++r)
      tileT[wv][(r + half * 8) * 16 + col] = acc[r];
    asm volatile("s_wait_dscnt 0" ::: "memory");
#pragma unroll
    for (int j = 0; j < 16; ++j) {
      int n = n0 + j;
      if (n < Vv) {
        float nv = fmaxf(tileT[wv][col * 16 + j], 0.0f);
        int ni = n;
#pragma unroll
        for (int t = 0; t < 5; ++t) {
          bool g = nv > topv[t];
          float ov = topv[t]; int oi = topi[t];
          topv[t] = g ? nv : topv[t];
          topi[t] = g ? ni : topi[t];
          nv = g ? ov : nv;
          ni = g ? oi : ni;
        }
      }
    }
  }

  const float peak = fmaxf(topv[0], 1e-6f);
  const float thr  = 0.25f * peak;
  const float mval = mask[p0 + col];

  // ---- pass 2: pred+target projections; sf / nonneg / peak terms ----
  float sf_num = 0, sf_den = 0, nonneg = 0, pk_num = 0, pk_cnt = 0;
  for (int nt = 0; nt < NT; ++nt) {
    const int n0 = nt * 16;
    v8f ap = {};
    v8f at = {};
#pragma unroll
    for (int s = 0; s < 2; ++s) {
      const __bf16* bsrc = &sfB[(n0 + col) * KP + s * 32 + half * 16];
      v16bf bf = *(const v16bf*)bsrc;
      ap = __builtin_amdgcn_wmma_f32_16x16x32_bf16(
          false, aP[s], false, bf, (short)0, ap, false, false);
      at = __builtin_amdgcn_wmma_f32_16x16x32_bf16(
          false, aT[s], false, bf, (short)0, at, false, false);
    }
#pragma unroll
    for (int r = 0; r < 8; ++r) {
      tileP[wv][(r + half * 8) * 16 + col] = ap[r];
      tileT[wv][(r + half * 8) * 16 + col] = at[r];
    }
    asm volatile("s_wait_dscnt 0" ::: "memory");
#pragma unroll
    for (int j = 0; j < 8; ++j) {
      int jj = j + half * 8;   // halves split columns: no double counting
      int n = n0 + jj;
      if (n < Vv) {
        float ps = tileP[wv][col * 16 + jj];
        float ts = tileT[wv][col * 16 + jj];
        float tpos  = fmaxf(ts, 0.0f);
        float tnorm = tpos / peak;
        float wgt = (1.0f + 4.0f * tnorm * tnorm) * mval;
        float r = ps - ts;
        sf_num += sqrtf(r * r + EPSf * EPSf) * wgt;
        sf_den += wgt;
        nonneg += fmaxf(-ps, 0.0f) * mval;
#pragma unroll
        for (int t = 0; t < 5; ++t) {
          if (n == topi[t]) {   // pred_peak == pred_sf gathered at top index
            float valid = ((topv[t] > thr) && (mval > 0.5f)) ? 1.0f : 0.0f;
            float d = ps - topv[t];
            pk_num += sqrtf(d * d + EPSf * EPSf) * valid;
            pk_cnt += valid;
          }
        }
      }
    }
  }

  // ---- per-pixel scalar terms (one lane per pixel) ----
  float fodf_c = 0, corr_c = 0, pow_c = 0, mask_c = 0;
  if (lane < 16) {
    fodf_c = cbsum * mval;
    const float inv = 1.0f / (float)Cc;
    float pmtm = ppt - psum * tsum * inv;
    float pm2  = fmaxf(ppv - psum * psum * inv, 0.0f);
    float tm2  = fmaxf(ttv - tsum * tsum * inv, 0.0f);
    float corr = pmtm / (sqrtf(pm2) * sqrtf(tm2) + 1e-6f);
    corr_c = (1.0f - corr) * mval;
    float pwv = 0.0f;
#pragma unroll
    for (int q = 0; q < 5; ++q) {
      float d = sqrtf(bp[q] + EPSf * EPSf) - sqrtf(bt[q] + EPSf * EPSf);
      pwv += sqrtf(d * d + EPSf * EPSf);
    }
    pow_c = pwv * mval;
    mask_c = mval;
  }

  // ---- wave reduction + global atomics ----
  float acc9[9] = {fodf_c, corr_c, sf_num, sf_den, nonneg, pk_num, pk_cnt, pow_c, mask_c};
#pragma unroll
  for (int i = 0; i < 9; ++i) {
    float v = acc9[i];
    v += __shfl_xor(v, 1);
    v += __shfl_xor(v, 2);
    v += __shfl_xor(v, 4);
    v += __shfl_xor(v, 8);
    v += __shfl_xor(v, 16);
    if (lane == 0) atomicAdd(&ws[i], v);
  }
}

__global__ void fodf_loss_finalize(const float* __restrict__ ws, float* __restrict__ out) {
  if (threadIdx.x != 0) return;
  float msum = ws[8];
  float fodf = ws[0] / fmaxf(msum * (float)Cc, 1.0f);
  float corr = ws[1] / fmaxf(msum, 1.0f);
  float sfl  = ws[2] / fmaxf(ws[3], 1.0f);
  float nn   = ws[4] / fmaxf(msum * (float)Vv, 1.0f);
  float pk   = ws[5] / fmaxf(ws[6], 1.0f);
  float pwl  = ws[7] / fmaxf(msum * 5.0f, 1.0f);
  out[0] = 1.0f * fodf + 0.1f * corr + 1.0f * sfl + 0.5f * pk + 0.1f * nn + 0.1f * pwl;
}

extern "C" void kernel_launch(void* const* d_in, const int* in_sizes, int n_in,
                              void* d_out, int out_size, void* d_ws, size_t ws_size,
                              hipStream_t stream) {
  (void)in_sizes; (void)n_in; (void)out_size; (void)ws_size;
  const float* pred = (const float*)d_in[0];
  const float* targ = (const float*)d_in[1];
  const float* sf   = (const float*)d_in[2];
  const float* mask = (const float*)d_in[3];
  float* ws  = (float*)d_ws;
  float* out = (float*)d_out;

  fodf_loss_zero<<<1, 32, 0, stream>>>(ws);
  fodf_loss_main<<<TILES / WAVES, WAVES * 32, 0, stream>>>(pred, targ, sf, mask, ws);
  fodf_loss_finalize<<<1, 32, 0, stream>>>(ws, out);
}
